// SelfAttention_80315888436028
// MI455X (gfx1250) — matrix operand order repaired and checked by reading, not compiled, not run on hardware
//
#include <hip/hip_runtime.h>
#include <hip/hip_bf16.h>

// ---------------------------------------------------------------------------
// Self-attention block for MI455X (gfx1250, wave32, WMMA).
//   x:(1,3072,1536) f32; Wq/Wk/Wv/Wo:(1536,1536); 12 heads, d=128, RoPE 3D grid.
// All GEMMs & attention einsums run on v_wmma_f32_16x16x32_bf16 (f32 accum).
// Flash-attention streams K/V; K tiles are double-buffered in LDS via
// global_load_async_to_lds_b128 (ASYNCcnt) to overlap fetch with compute.
// ---------------------------------------------------------------------------

typedef __attribute__((ext_vector_type(16))) __bf16 v16bf;
typedef __attribute__((ext_vector_type(8)))  float  v8f;

union FragB {
    v16bf          v;
    uint4          q[2];
    unsigned short us[16];
};

__device__ __forceinline__ unsigned short f2bf(float f) {
    unsigned int u = __float_as_uint(f);
    u += 0x7FFFu + ((u >> 16) & 1u);          // round-to-nearest-even
    return (unsigned short)(u >> 16);
}

// ---------------------------------------------------------------------------
// f32 -> bf16 conversion
// ---------------------------------------------------------------------------
__global__ __launch_bounds__(256)
void cvt_bf16_kernel(const float* __restrict__ in, unsigned short* __restrict__ out, int n) {
    int i = blockIdx.x * 256 + threadIdx.x;
    if (i < n) out[i] = f2bf(in[i]);
}

// ---------------------------------------------------------------------------
// C(MxN,f32) = A(MxK,bf16) * B(NxK,bf16)^T + bias(N)
// block = 128 threads = 4 waves (2x2), 64x128 tile, wave = 32x64 (2x4 frags)
// 8 WMMAs per 6 fragment loads per k-step.
// ---------------------------------------------------------------------------
__global__ __launch_bounds__(128)
void gemm_bf16_nt(const unsigned short* __restrict__ A,
                  const unsigned short* __restrict__ B,
                  const float* __restrict__ bias,
                  float* __restrict__ C,
                  int M, int N, int K)
{
    const int wave = threadIdx.x >> 5;
    const int lane = threadIdx.x & 31;
    const int half = lane >> 4;
    const int l16  = lane & 15;
    const int m0   = blockIdx.y * 64  + (wave >> 1) * 32;
    const int n0   = blockIdx.x * 128 + (wave &  1) * 64;

    v8f acc[2][4] = {};

#pragma unroll 2
    for (int k0 = 0; k0 < K; k0 += 32) {
        FragB a[2], b[4];
#pragma unroll
        for (int mi = 0; mi < 2; ++mi) {
            const unsigned short* ap = A + (size_t)(m0 + mi * 16 + l16) * K + k0 + half * 8;
            a[mi].q[0] = *(const uint4*)(ap);
            a[mi].q[1] = *(const uint4*)(ap + 16);
        }
#pragma unroll
        for (int ni = 0; ni < 4; ++ni) {
            const unsigned short* bp = B + (size_t)(n0 + ni * 16 + l16) * K + k0 + half * 8;
            b[ni].q[0] = *(const uint4*)(bp);
            b[ni].q[1] = *(const uint4*)(bp + 16);
        }
#pragma unroll
        for (int mi = 0; mi < 2; ++mi)
#pragma unroll
            for (int ni = 0; ni < 4; ++ni)
                acc[mi][ni] = __builtin_amdgcn_wmma_f32_16x16x32_bf16(
                    false, a[mi].v, false, b[ni].v, (short)0, acc[mi][ni], false, false);
    }

#pragma unroll
    for (int mi = 0; mi < 2; ++mi)
#pragma unroll
        for (int ni = 0; ni < 4; ++ni) {
            const int col = n0 + ni * 16 + l16;
            const float bv = bias[col];
#pragma unroll
            for (int j = 0; j < 8; ++j) {
                const int row = m0 + mi * 16 + half * 8 + j;
                C[(size_t)row * N + col] = acc[mi][ni][j] + bv;
            }
        }
}

// ---------------------------------------------------------------------------
// Epilogue: mode 0 -> rmsnorm(row,1536)*g then RoPE per head, cast bf16
//           mode 1 -> cast bf16 only (V path)
// grid = 3072 rows, 256 threads
// ---------------------------------------------------------------------------
__global__ __launch_bounds__(256)
void qkv_epilogue(const float* __restrict__ in,
                  const float* __restrict__ g,
                  const float* __restrict__ fcos,
                  const float* __restrict__ fsin,
                  unsigned short* __restrict__ outb,
                  int mode)
{
    const int s   = blockIdx.x;
    const int tid = threadIdx.x;
    const float* row = in + (size_t)s * 1536;

    __shared__ float red[8];
    float inv = 1.0f;
    if (mode == 0) {
        float ss = 0.f;
        for (int c = tid; c < 1536; c += 256) { float t = row[c]; ss += t * t; }
#pragma unroll
        for (int m = 1; m < 32; m <<= 1) ss += __shfl_xor(ss, m, 32);
        if ((tid & 31) == 0) red[tid >> 5] = ss;
        __syncthreads();
        float tot = 0.f;
#pragma unroll
        for (int w = 0; w < 8; ++w) tot += red[w];
        inv = rsqrtf(tot * (1.0f / 1536.0f) + 1e-6f);
    }

    // position -> (f,h,w) indices of the 3x32x32 RoPE grid
    const int fidx = s >> 10;
    const int rem  = s & 1023;
    const int hidx = rem >> 5;
    const int widx = rem & 31;

    for (int p = tid; p < 768; p += 256) {      // pair index over dim=1536
        const int c  = p * 2;
        float xr = row[c], xi = row[c + 1];
        if (mode == 0) {
            xr *= inv * g[c];
            xi *= inv * g[c + 1];
            const int j    = (c & 127) >> 1;    // 0..63 pair index within head
            const int rrow = (j < 22) ? fidx : ((j < 43) ? hidx : widx);
            const float cs = fcos[rrow * 64 + j];
            const float sn = fsin[rrow * 64 + j];
            const float orr = xr * cs - xi * sn;
            const float oii = xr * sn + xi * cs;
            xr = orr; xi = oii;
        }
        outb[(size_t)s * 1536 + c]     = f2bf(xr);
        outb[(size_t)s * 1536 + c + 1] = f2bf(xi);
    }
}

// ---------------------------------------------------------------------------
// Flash attention: grid = (48 query-tiles, 12 heads), 128 thr = 4 waves.
// Each wave owns 16 query rows. K tiles double-buffered in LDS via
// global_load_async_to_lds_b128; V staged transposed so PV B-fragments are
// contiguous. Online softmax across 16-lane C-matrix halves.
// ---------------------------------------------------------------------------
__global__ __launch_bounds__(128)
void flash_attn(const unsigned short* __restrict__ Q,
                const unsigned short* __restrict__ Kc,
                const unsigned short* __restrict__ V,
                unsigned short* __restrict__ O)
{
    const int S = 3072, DIMT = 1536, D = 128;
    const float scale = 0.08838834764831845f;   // 1/sqrt(128)

    const int q0   = blockIdx.x * 64;
    const int head = blockIdx.y;
    const int tid  = threadIdx.x;
    const int wave = tid >> 5;
    const int lane = tid & 31;
    const int half = lane >> 4;
    const int l16  = lane & 15;

    __shared__ unsigned short Ks[2][64 * 128];  // [buf][t][d], async-filled
    __shared__ unsigned short Vs[128 * 64];     // transposed: [d][t]
    __shared__ unsigned short Ps[4][16 * 64];   // per-wave P staging

    // Issue async K-tile copy (16 KB) : 8 x B128 per thread-slot, ASYNCcnt-tracked
    auto issueK = [&](int k0, int buf) {
#pragma unroll
        for (int it = 0; it < 8; ++it) {
            const int idx  = tid + it * 128;
            const int row  = idx >> 4;
            const int col8 = (idx & 15) * 8;
            // addrspace(3) pointer value: low 32 bits are the LDS offset
            unsigned laddr = (unsigned)(unsigned long long)&Ks[buf][row * 128 + col8];
            const unsigned short* gp = Kc + (size_t)(k0 + row) * DIMT + head * D + col8;
            asm volatile("global_load_async_to_lds_b128 %0, %1, off"
                         :: "v"(laddr), "v"(gp) : "memory");
        }
    };

    // Load this wave's Q rows as A-fragments (contraction over d=128, 4 steps)
    FragB qa[4];
    const int qrow = q0 + wave * 16 + l16;
#pragma unroll
    for (int kkI = 0; kkI < 4; ++kkI) {
        const unsigned short* qp = Q + (size_t)qrow * DIMT + head * D + kkI * 32 + half * 8;
        qa[kkI].q[0] = *(const uint4*)qp;
        qa[kkI].q[1] = *(const uint4*)(qp + 16);
    }

    v8f oacc[8] = {};
    float mrow[8], lrow[8];
#pragma unroll
    for (int j = 0; j < 8; ++j) { mrow[j] = -1e30f; lrow[j] = 0.f; }

    issueK(0, 0);                               // prefetch tile 0

    for (int t = 0; t < S / 64; ++t) {
        const int k0  = t * 64;
        const int cur = t & 1;

        asm volatile("s_wait_asynccnt 0x0" ::: "memory");   // Ks[cur] landed
        __syncthreads();                         // all waves: Ks[cur] ready, Vs free

        if (t + 1 < S / 64) issueK(k0 + 64, cur ^ 1);       // overlap next K fetch

        // ---- stage V tile transposed (sync) ----
#pragma unroll
        for (int it = 0; it < 8; ++it) {
            const int idx  = tid + it * 128;
            const int row  = idx >> 4;
            const int col8 = (idx & 15) * 8;
            uint4 d4 = *(const uint4*)(V + (size_t)(k0 + row) * DIMT + head * D + col8);
            const unsigned short* e = (const unsigned short*)&d4;
#pragma unroll
            for (int b = 0; b < 8; ++b) Vs[(col8 + b) * 64 + row] = e[b];
        }
        __syncthreads();

        // ---- scores: S = Q K^T  (16x64 per wave, 16 WMMA) ----
        v8f sc[4] = {};
#pragma unroll
        for (int kkI = 0; kkI < 4; ++kkI) {
#pragma unroll
            for (int ci = 0; ci < 4; ++ci) {
                FragB kb_;
                const unsigned short* kp = &Ks[cur][(ci * 16 + l16) * 128 + kkI * 32 + half * 8];
                kb_.q[0] = *(const uint4*)kp;
                kb_.q[1] = *(const uint4*)(kp + 16);
                sc[ci] = __builtin_amdgcn_wmma_f32_16x16x32_bf16(
                    false, qa[kkI].v, false, kb_.v, (short)0, sc[ci], false, false);
            }
        }
#pragma unroll
        for (int ci = 0; ci < 4; ++ci)
#pragma unroll
            for (int j = 0; j < 8; ++j) sc[ci][j] *= scale;

        // ---- online softmax (rows live across 16-lane halves) ----
        float alpha[8];
#pragma unroll
        for (int j = 0; j < 8; ++j) {
            float mx = fmaxf(fmaxf(sc[0][j], sc[1][j]), fmaxf(sc[2][j], sc[3][j]));
            mx = fmaxf(mx, __shfl_xor(mx, 1, 32));
            mx = fmaxf(mx, __shfl_xor(mx, 2, 32));
            mx = fmaxf(mx, __shfl_xor(mx, 4, 32));
            mx = fmaxf(mx, __shfl_xor(mx, 8, 32));
            const float mn = fmaxf(mrow[j], mx);
            alpha[j] = __expf(mrow[j] - mn);
            mrow[j]  = mn;
        }
        float rs[8];
#pragma unroll
        for (int j = 0; j < 8; ++j) rs[j] = 0.f;
#pragma unroll
        for (int ci = 0; ci < 4; ++ci)
#pragma unroll
            for (int j = 0; j < 8; ++j) {
                const float p = __expf(sc[ci][j] - mrow[j]);
                sc[ci][j] = p;
                rs[j] += p;
            }
#pragma unroll
        for (int j = 0; j < 8; ++j) {
            rs[j] += __shfl_xor(rs[j], 1, 32);
            rs[j] += __shfl_xor(rs[j], 2, 32);
            rs[j] += __shfl_xor(rs[j], 4, 32);
            rs[j] += __shfl_xor(rs[j], 8, 32);
            lrow[j] = lrow[j] * alpha[j] + rs[j];
        }
#pragma unroll
        for (int ni = 0; ni < 8; ++ni)
#pragma unroll
            for (int j = 0; j < 8; ++j) oacc[ni][j] *= alpha[j];

        // ---- P: C-layout -> A-layout via per-wave LDS staging ----
        unsigned short* pw = &Ps[wave][0];
#pragma unroll
        for (int ci = 0; ci < 4; ++ci)
#pragma unroll
            for (int j = 0; j < 8; ++j)
                pw[(half * 8 + j) * 64 + ci * 16 + l16] = f2bf(sc[ci][j]);
        __syncthreads();

        // ---- O += P V  (contraction over 64 keys, 16 WMMA) ----
#pragma unroll
        for (int kk2 = 0; kk2 < 2; ++kk2) {
            FragB pa;
            const unsigned short* pp = &Ps[wave][l16 * 64 + kk2 * 32 + half * 8];
            pa.q[0] = *(const uint4*)pp;
            pa.q[1] = *(const uint4*)(pp + 16);
#pragma unroll
            for (int ni = 0; ni < 8; ++ni) {
                FragB vb_;
                const unsigned short* vp = &Vs[(ni * 16 + l16) * 64 + kk2 * 32 + half * 8];
                vb_.q[0] = *(const uint4*)vp;
                vb_.q[1] = *(const uint4*)(vp + 16);
                oacc[ni] = __builtin_amdgcn_wmma_f32_16x16x32_bf16(
                    false, pa.v, false, vb_.v, (short)0, oacc[ni], false, false);
            }
        }
    }

    // ---- normalize + write O (bf16, row-major 3072x1536) ----
#pragma unroll
    for (int ni = 0; ni < 8; ++ni)
#pragma unroll
        for (int j = 0; j < 8; ++j) {
            const int row = q0 + wave * 16 + half * 8 + j;
            const int d   = ni * 16 + l16;
            O[(size_t)row * DIMT + head * D + d] = f2bf(oacc[ni][j] / lrow[j]);
        }
}

// ---------------------------------------------------------------------------
extern "C" void kernel_launch(void* const* d_in, const int* in_sizes, int n_in,
                              void* d_out, int out_size, void* d_ws, size_t ws_size,
                              hipStream_t stream)
{
    (void)in_sizes; (void)n_in; (void)out_size; (void)ws_size;
    const int S = 3072, DIM = 1536;
    const size_t SZ_X = (size_t)S * DIM;       // 4718592
    const size_t SZ_W = (size_t)DIM * DIM;     // 2359296

    const float* x    = (const float*)d_in[0];
    const float* wq   = (const float*)d_in[1];
    const float* bq   = (const float*)d_in[2];
    const float* wk   = (const float*)d_in[3];
    const float* bk   = (const float*)d_in[4];
    const float* wv   = (const float*)d_in[5];
    const float* bv   = (const float*)d_in[6];
    const float* wo   = (const float*)d_in[7];
    const float* bo   = (const float*)d_in[8];
    const float* gq   = (const float*)d_in[9];
    const float* gk   = (const float*)d_in[10];
    const float* fcos = (const float*)d_in[11];
    const float* fsin = (const float*)d_in[12];

    char* w = (char*)d_ws;
    size_t off = 0;
    unsigned short* xb  = (unsigned short*)(w + off); off += SZ_X * 2;
    unsigned short* wqb = (unsigned short*)(w + off); off += SZ_W * 2;
    unsigned short* wkb = (unsigned short*)(w + off); off += SZ_W * 2;
    unsigned short* wvb = (unsigned short*)(w + off); off += SZ_W * 2;
    unsigned short* wob = (unsigned short*)(w + off); off += SZ_W * 2;
    float*          qf  = (float*)(w + off);
    size_t qf_off = off;                              off += SZ_X * 4;
    float*          kf  = (float*)(w + off);          off += SZ_X * 4;
    float*          vf  = (float*)(w + off);          off += SZ_X * 4;
    unsigned short* qbb = (unsigned short*)(w + off); off += SZ_X * 2;
    unsigned short* kbb = (unsigned short*)(w + off); off += SZ_X * 2;
    unsigned short* vbb = (unsigned short*)(w + off); off += SZ_X * 2;
    unsigned short* obb = (unsigned short*)(w + qf_off);  // alias: qf dead after epilogue

    // 1) f32 -> bf16 conversions
    cvt_bf16_kernel<<<(int)((SZ_X + 255) / 256), 256, 0, stream>>>(x,  xb,  (int)SZ_X);
    cvt_bf16_kernel<<<(int)((SZ_W + 255) / 256), 256, 0, stream>>>(wq, wqb, (int)SZ_W);
    cvt_bf16_kernel<<<(int)((SZ_W + 255) / 256), 256, 0, stream>>>(wk, wkb, (int)SZ_W);
    cvt_bf16_kernel<<<(int)((SZ_W + 255) / 256), 256, 0, stream>>>(wv, wvb, (int)SZ_W);
    cvt_bf16_kernel<<<(int)((SZ_W + 255) / 256), 256, 0, stream>>>(wo, wob, (int)SZ_W);

    // 2) QKV projections (WMMA bf16, f32 accum, bias fused)
    dim3 gg(DIM / 128, S / 64);
    gemm_bf16_nt<<<gg, 128, 0, stream>>>(xb, wqb, bq, qf, S, DIM, DIM);
    gemm_bf16_nt<<<gg, 128, 0, stream>>>(xb, wkb, bk, kf, S, DIM, DIM);
    gemm_bf16_nt<<<gg, 128, 0, stream>>>(xb, wvb, bv, vf, S, DIM, DIM);

    // 3) rmsnorm + RoPE (q,k) / cast (v) -> bf16
    qkv_epilogue<<<S, 256, 0, stream>>>(qf, gq, fcos, fsin, qbb, 0);
    qkv_epilogue<<<S, 256, 0, stream>>>(kf, gk, fcos, fsin, kbb, 0);
    qkv_epilogue<<<S, 256, 0, stream>>>(vf, gq, fcos, fsin, vbb, 1);

    // 4) flash attention (12 heads x 48 query tiles)
    flash_attn<<<dim3(S / 64, 12), 128, 0, stream>>>(qbb, kbb, vbb, obb);

    // 5) output projection -> d_out (f32)
    gemm_bf16_nt<<<gg, 128, 0, stream>>>(obb, wob, bo, (float*)d_out, S, DIM, DIM);
}
